// texure_point_net_20950850469947
// MI455X (gfx1250) — compile-verified, matrix operand order checked by reading, not run on hardware
//
#include <hip/hip_runtime.h>
#include <hip/hip_bf16.h>
#include <math.h>

// ---------------- problem constants (match reference) ----------------
#define NPTS  16384
#define NCLS  13
#define NS1   2458     // ceil(16384*0.15)
#define NS2   369      // ceil(2458*0.15)

// ---------------- WMMA / vector types (gfx1250, wave32) ----------------
typedef __attribute__((ext_vector_type(16))) __bf16 v16bf;
typedef __attribute__((ext_vector_type(8)))  __bf16 bf16x8;
typedef __attribute__((ext_vector_type(8)))  float  v8f;
typedef __attribute__((ext_vector_type(4)))  float  f32x4;

__device__ __forceinline__ float sqf(float x){ return x*x; }

// =====================================================================
// Texture module: 3x12x12 atlas -> conv3x3(16)+affine/relu+pool2
// -> conv2x2(32)+affine/relu+pool2 -> linear(128->32).
// One workgroup (128 threads) per point. Also copies 49 raw features.
// =====================================================================
__global__ void tex_kernel(const float* __restrict__ atlas, const float* __restrict__ feat,
                           const float* __restrict__ W1,const float* __restrict__ b1,
                           const float* __restrict__ s1,const float* __restrict__ h1,
                           const float* __restrict__ W2,const float* __restrict__ b2,
                           const float* __restrict__ s2,const float* __restrict__ h2,
                           const float* __restrict__ Wl,const float* __restrict__ bl,
                           float* __restrict__ x0){
  __shared__ float sa[432];
  __shared__ float p1[400];
  __shared__ float p2[128];
  const int pt = blockIdx.x, tid = threadIdx.x, nt = blockDim.x;
  const float* ap = atlas + (size_t)pt*432;
  for(int i=tid;i<432;i+=nt) sa[i]=ap[i];
  __syncthreads();
  for(int o=tid;o<400;o+=nt){
    int c=o/25, py=(o%25)/5, px=o%5;
    float mx=-3.4e38f;
    for(int dy=0;dy<2;dy++) for(int dx=0;dx<2;dx++){
      int iy=py*2+dy, ix=px*2+dx;
      float v=0.f;
      for(int ic=0;ic<3;ic++)
        for(int ky=0;ky<3;ky++)
          for(int kx=0;kx<3;kx++)
            v += sa[ic*144+(iy+ky)*12+(ix+kx)] * W1[((c*3+ic)*3+ky)*3+kx];
      v=(v+b1[c])*s1[c]+h1[c];
      v=v>0.f?v:0.f;
      mx=v>mx?v:mx;
    }
    p1[o]=mx;
  }
  __syncthreads();
  for(int o=tid;o<128;o+=nt){
    int c=o/4, py=(o%4)/2, px=o%2;
    float mx=-3.4e38f;
    for(int dy=0;dy<2;dy++) for(int dx=0;dx<2;dx++){
      int iy=py*2+dy, ix=px*2+dx;
      float v=0.f;
      for(int ic=0;ic<16;ic++)
        for(int ky=0;ky<2;ky++)
          for(int kx=0;kx<2;kx++)
            v += p1[ic*25+(iy+ky)*5+(ix+kx)] * W2[((c*16+ic)*2+ky)*2+kx];
      v=(v+b2[c])*s2[c]+h2[c];
      v=v>0.f?v:0.f;
      mx=v>mx?v:mx;
    }
    p2[o]=mx;
  }
  __syncthreads();
  float* xr = x0 + (size_t)pt*81;
  for(int o=tid;o<32;o+=nt){
    float v=bl[o];
    for(int i=0;i<128;i++) v += p2[i]*Wl[i*32+o];
    xr[o]=v;
  }
  const float* fr = feat + (size_t)pt*49;
  for(int o=tid;o<49;o+=nt) xr[32+o]=fr[o];
}

// =====================================================================
// Farthest point sampling (start at 0, first-max tiebreak). Single block.
// =====================================================================
__global__ void fps_kernel(const float* __restrict__ pos, int n, int nsamp,
                           int* __restrict__ idx, float* __restrict__ d){
  __shared__ float rv[1024];
  __shared__ int   ri[1024];
  __shared__ int   sb;
  const int tid=threadIdx.x, nt=blockDim.x;
  float p0x=pos[0],p0y=pos[1],p0z=pos[2];
  for(int i=tid;i<n;i+=nt){
    const float* p=pos+3*i;
    d[i]=sqf(p[0]-p0x)+sqf(p[1]-p0y)+sqf(p[2]-p0z);
  }
  if(tid==0) idx[0]=0;
  __syncthreads();
  for(int s=1;s<nsamp;s++){
    float bv=-1.f; int bi=0x7fffffff;
    for(int i=tid;i<n;i+=nt){ float v=d[i]; if(v>bv){bv=v;bi=i;} }
    rv[tid]=bv; ri[tid]=bi; __syncthreads();
    for(int off=nt>>1;off>0;off>>=1){
      if(tid<off){
        float ov=rv[tid+off]; int oi=ri[tid+off];
        if(ov>rv[tid] || (ov==rv[tid] && oi<ri[tid])){ rv[tid]=ov; ri[tid]=oi; }
      }
      __syncthreads();
    }
    if(tid==0){ idx[s]=ri[0]; sb=ri[0]; }
    __syncthreads();
    const float* pb=pos+3*sb;
    float bx=pb[0],by=pb[1],bz=pb[2];
    for(int i=tid;i<n;i+=nt){
      const float* p=pos+3*i;
      float nd=sqf(p[0]-bx)+sqf(p[1]-by)+sqf(p[2]-bz);
      d[i]=d[i]<nd?d[i]:nd;
    }
    __syncthreads();
  }
}

__global__ void gather3(const float* __restrict__ pos, const int* __restrict__ idx,
                        int n, float* __restrict__ out){
  int i=blockIdx.x*blockDim.x+threadIdx.x;
  if(i<n*3) out[i]=pos[(size_t)idx[i/3]*3 + (i%3)];
}

// =====================================================================
// KNN large-k: one block/query, distances in dynamic LDS, k min-extractions.
// =====================================================================
__global__ void knn_block(const float* __restrict__ src, int nsrc,
                          const float* __restrict__ qp, int k, int* __restrict__ nbr){
  extern __shared__ float smem[];
  float* sd = smem;
  float* rv = smem + nsrc;
  int*   ri = (int*)(rv + blockDim.x);
  const int q=blockIdx.x, tid=threadIdx.x, nt=blockDim.x;
  float qx=qp[q*3],qy=qp[q*3+1],qz=qp[q*3+2];
  for(int i=tid;i<nsrc;i+=nt){
    const float* p=src+3*i;
    sd[i]=sqf(p[0]-qx)+sqf(p[1]-qy)+sqf(p[2]-qz);
  }
  __syncthreads();
  for(int t=0;t<k;t++){
    float bv=3.5e38f; int bi=0x7fffffff;
    for(int i=tid;i<nsrc;i+=nt){ float v=sd[i]; if(v<bv){bv=v;bi=i;} }
    rv[tid]=bv; ri[tid]=bi; __syncthreads();
    for(int off=nt>>1;off>0;off>>=1){
      if(tid<off){
        float ov=rv[tid+off]; int oi=ri[tid+off];
        if(ov<rv[tid] || (ov==rv[tid] && oi<ri[tid])){ rv[tid]=ov; ri[tid]=oi; }
      }
      __syncthreads();
    }
    if(tid==0){ nbr[(size_t)q*k+t]=ri[0]; sd[ri[0]]=3.4e38f; }
    __syncthreads();
  }
}

// KNN k=3: one thread per query.
__global__ void knn3_kernel(const float* __restrict__ src, int nsrc,
                            const float* __restrict__ qp, int nq,
                            int* __restrict__ idx, float* __restrict__ d2o){
  int q=blockIdx.x*blockDim.x+threadIdx.x;
  if(q>=nq) return;
  float qx=qp[q*3],qy=qp[q*3+1],qz=qp[q*3+2];
  float bd0=3.4e38f,bd1=3.4e38f,bd2=3.4e38f;
  int   bi0=0,bi1=0,bi2=0;
  for(int i=0;i<nsrc;i++){
    const float* p=src+3*i;
    float v=sqf(p[0]-qx)+sqf(p[1]-qy)+sqf(p[2]-qz);
    if(v<bd0){ bd2=bd1;bi2=bi1; bd1=bd0;bi1=bi0; bd0=v;bi0=i; }
    else if(v<bd1){ bd2=bd1;bi2=bi1; bd1=v;bi1=i; }
    else if(v<bd2){ bd2=v;bi2=i; }
  }
  idx[q*3+0]=bi0; idx[q*3+1]=bi1; idx[q*3+2]=bi2;
  d2o[q*3+0]=bd0; d2o[q*3+1]=bd1; d2o[q*3+2]=bd2;
}

__device__ __forceinline__ unsigned hash_u32(unsigned x){
  x^=x>>16; x*=0x7feb352dU; x^=x>>15; x*=0x846ca68bU; x^=x>>16; return x;
}
__global__ void subsample_kernel(const int* __restrict__ nbrF, int nq, int K, int KD,
                                 unsigned seed, int* __restrict__ nbrS){
  int t=blockIdx.x*blockDim.x+threadIdx.x;
  if(t>=nq*K) return;
  int q=t/K;
  unsigned r=hash_u32((unsigned)t*2654435761U ^ seed);
  nbrS[t]=nbrF[(size_t)q*KD + (r%KD)];
}

// msg[r=q*K+j, :] = concat(x[nbr[r]], pos[nbr[r]] - qpos[q]); zero-pad to Wp cols.
__global__ void build_msg(const float* __restrict__ x, int F, int ldx,
                          const float* __restrict__ pos, const float* __restrict__ qp,
                          const int* __restrict__ nbr, int K, float* __restrict__ msg, int Wp){
  int r=blockIdx.x;
  int q=r/K;
  int nb=nbr[r];
  float* mr=msg+(size_t)r*Wp;
  const float* xr=x+(size_t)nb*ldx;
  for(int c=threadIdx.x;c<Wp;c+=blockDim.x){
    float v=0.f;
    if(c<F) v=xr[c];
    else if(c<F+3) v=pos[(size_t)nb*3+(c-F)] - qp[(size_t)q*3+(c-F)];
    mr[c]=v;
  }
}

// =====================================================================
// Weight prep: f32 [K,N] row-major  ->  bf16 [Np][Kp] (transposed, padded,
// Np%32==0, Kp%32==0, zero-filled pads).
// =====================================================================
__global__ void prep_w(const float* __restrict__ W, int K, int N,
                       __bf16* __restrict__ out, int Kp, int Np){
  int t=blockIdx.x*blockDim.x+threadIdx.x;
  if(t>=Np*Kp) return;
  int n=t/Kp, kk=t%Kp;
  float v=(n<N && kk<K)? W[(size_t)kk*N+n] : 0.f;
  out[t]=(__bf16)v;
}

// =====================================================================
// GEMM via V_WMMA_F32_16X16X32_BF16, 2x2 tile blocking: each wave32
// computes a 32x32 output block (4 accumulators), halving redundant
// A (f32) and B (bf16) traffic vs one-tile-per-wave.
// A: f32 [.., Kp] lda (Kp%32==0, pad cols zeroed by producers; row
// overhang up to 32 stays inside the scratch arenas).
// B: bf16 [Np][Kp] (prep_w output, Np%32==0).
// K loop is branch-free: 8x float4 A loads + 4x bf16x8 B loads + cvt +
// 4x wmma per K-step. Fused epilogue: +bias, *s+h, +Cadd, ReLU; masked.
// =====================================================================
__global__ void gemm_wmma(const float* __restrict__ A, const __bf16* __restrict__ Wb,
                          const float* __restrict__ bias, const float* __restrict__ sc,
                          const float* __restrict__ sh, const float* __restrict__ Cadd,
                          float* __restrict__ C,
                          int M, int N, int Kp, int lda, int ldc, int relu){
  const int tilesN=(N+31)>>5;
  const int tilesM=(M+31)>>5;
  const int wid = blockIdx.x*(blockDim.x>>5) + (threadIdx.x>>5);
  if(wid>=tilesM*tilesN) return;           // wave-uniform exit
  const int tm=wid/tilesN, tn=wid%tilesN;
  const int lane=threadIdx.x&31;
  const int hf=lane>>4;
  const int r=lane&15;
  const float*  Ar0 = A  + (size_t)(tm*32+r)*lda + 8*hf;
  const float*  Ar1 = Ar0 + (size_t)16*lda;
  const __bf16* Bc0 = Wb + (size_t)(tn*32+r)*Kp + 8*hf;
  const __bf16* Bc1 = Bc0 + (size_t)16*Kp;
  v8f acc00={0.f,0.f,0.f,0.f,0.f,0.f,0.f,0.f};
  v8f acc01=acc00, acc10=acc00, acc11=acc00;
  for(int k0=0;k0<Kp;k0+=32){
    f32x4 a00=*(const f32x4*)(Ar0+k0);
    f32x4 a01=*(const f32x4*)(Ar0+k0+4);
    f32x4 a02=*(const f32x4*)(Ar0+k0+16);
    f32x4 a03=*(const f32x4*)(Ar0+k0+20);
    f32x4 a10=*(const f32x4*)(Ar1+k0);
    f32x4 a11=*(const f32x4*)(Ar1+k0+4);
    f32x4 a12=*(const f32x4*)(Ar1+k0+16);
    f32x4 a13=*(const f32x4*)(Ar1+k0+20);
    bf16x8 b00=*(const bf16x8*)(Bc0+k0);
    bf16x8 b01=*(const bf16x8*)(Bc0+k0+16);
    bf16x8 b10=*(const bf16x8*)(Bc1+k0);
    bf16x8 b11=*(const bf16x8*)(Bc1+k0+16);
    v16bf a0,a1,b0,b1;
#pragma unroll
    for(int e=0;e<4;e++){
      a0[e]   =(__bf16)a00[e];
      a0[4+e] =(__bf16)a01[e];
      a0[8+e] =(__bf16)a02[e];
      a0[12+e]=(__bf16)a03[e];
      a1[e]   =(__bf16)a10[e];
      a1[4+e] =(__bf16)a11[e];
      a1[8+e] =(__bf16)a12[e];
      a1[12+e]=(__bf16)a13[e];
    }
#pragma unroll
    for(int e=0;e<8;e++){ b0[e]=b00[e]; b0[8+e]=b01[e]; b1[e]=b10[e]; b1[8+e]=b11[e]; }
    acc00=__builtin_amdgcn_wmma_f32_16x16x32_bf16(false,a0,false,b0,(short)0,acc00,false,false);
    acc01=__builtin_amdgcn_wmma_f32_16x16x32_bf16(false,a0,false,b1,(short)0,acc01,false,false);
    acc10=__builtin_amdgcn_wmma_f32_16x16x32_bf16(false,a1,false,b0,(short)0,acc10,false,false);
    acc11=__builtin_amdgcn_wmma_f32_16x16x32_bf16(false,a1,false,b1,(short)0,acc11,false,false);
  }
  // epilogue: 4 sub-tiles, masked stores
#pragma unroll
  for(int t=0;t<4;t++){
    const v8f* ap = (t==0)?&acc00:(t==1)?&acc01:(t==2)?&acc10:&acc11;
    int mbase = tm*32 + ((t>>1)?16:0);
    int n     = tn*32 + ((t&1)?16:0) + r;
#pragma unroll
    for(int v=0;v<8;v++){
      int mm=mbase+v+8*hf;
      if(mm<M && n<N){
        float f=(*ap)[v];
        if(bias) f+=bias[n];
        if(sc)   f=f*sc[n]+sh[n];
        if(Cadd) f+=Cadd[(size_t)mm*ldc+n];
        if(relu) f=f>0.f?f:0.f;
        C[(size_t)mm*ldc+n]=f;
      }
    }
  }
}

// out[q, ofs+c] = max_j t[(q*K+j)*ldt + c]
__global__ void max_aggr(const float* __restrict__ t, int K, int C, int ldt,
                         float* __restrict__ out, int ldo, int ofs){
  int q=blockIdx.x;
  for(int c=threadIdx.x;c<C;c+=blockDim.x){
    float m=-3.4e38f;
    for(int j=0;j<K;j++){ float v=t[((size_t)q*K+j)*ldt+c]; m=v>m?v:m; }
    out[(size_t)q*ldo+ofs+c]=m;
  }
}

// out[r,:] = concat(a[r,:Ca], b[r,:Cb]); zero-pad to Wp.
__global__ void concat2(const float* __restrict__ a, int Ca,
                        const float* __restrict__ b, int Cb,
                        float* __restrict__ out, int Wp){
  int r=blockIdx.x;
  for(int c=threadIdx.x;c<Wp;c+=blockDim.x){
    float v=0.f;
    if(c<Ca) v=a[(size_t)r*Ca+c];
    else if(c<Ca+Cb) v=b[(size_t)r*Cb+(c-Ca)];
    out[(size_t)r*Wp+c]=v;
  }
}

// out[c] = max_r t[r,c]
__global__ void colmax(const float* __restrict__ t, int rows, int C, int ldt,
                       float* __restrict__ out){
  int c=blockIdx.x*blockDim.x+threadIdx.x;
  if(c>=C) return;
  float m=-3.4e38f;
  for(int r=0;r<rows;r++){ float v=t[(size_t)r*ldt+c]; m=v>m?v:m; }
  out[c]=m;
}

// knn_interpolate (inv-d2 weights, k<=3; k==0 -> broadcast xc row 0)
// concat skip; zero-pad to Wp.
__global__ void interp_concat(const float* __restrict__ xc, int Cc, int ldxc,
                              const int* __restrict__ idx, const float* __restrict__ d2, int k,
                              const float* __restrict__ skip, int Cs, int lds,
                              float* __restrict__ out, int Wp){
  int q=blockIdx.x;
  float w0=0.f,w1=0.f,w2=0.f,ws=1.f; int i0=0,i1=0,i2=0;
  if(k>0){
    float dd=d2[q*k+0]; dd=dd>1e-16f?dd:1e-16f; w0=1.f/dd; i0=idx[q*k+0];
    if(k>1){ dd=d2[q*k+1]; dd=dd>1e-16f?dd:1e-16f; w1=1.f/dd; i1=idx[q*k+1]; }
    if(k>2){ dd=d2[q*k+2]; dd=dd>1e-16f?dd:1e-16f; w2=1.f/dd; i2=idx[q*k+2]; }
    ws=w0+w1+w2;
  }
  for(int c=threadIdx.x;c<Wp;c+=blockDim.x){
    float v=0.f;
    if(c<Cc){
      if(k==0) v=xc[c];
      else{
        v=w0*xc[(size_t)i0*ldxc+c];
        if(k>1) v+=w1*xc[(size_t)i1*ldxc+c];
        if(k>2) v+=w2*xc[(size_t)i2*ldxc+c];
        v/=ws;
      }
    } else if(c<Cc+Cs) v=skip[(size_t)q*lds+(c-Cc)];
    out[(size_t)q*Wp+c]=v;
  }
}

__global__ void logsoftmax13(float* __restrict__ out, int n){
  int r=blockIdx.x*blockDim.x+threadIdx.x;
  if(r>=n) return;
  float* p=out+(size_t)r*NCLS;
  float m=p[0];
  for(int i=1;i<NCLS;i++) m=p[i]>m?p[i]:m;
  float s=0.f;
  for(int i=0;i<NCLS;i++) s+=expf(p[i]-m);
  float l=m+logf(s);
  for(int i=0;i<NCLS;i++) p[i]-=l;
}

// =====================================================================
// Host-side orchestration
// =====================================================================
struct LBN{ const float *W,*b,*s,*h; int K,N,Kp; __bf16* Wb; };
struct RES{ LBN l1,l2,dn; };

static inline int rup(int x,int a){ return (x+a-1)&~(a-1); }

static void gemm_launch(hipStream_t st, const float* A, const LBN& L,
                        const float* Cadd, float* C,
                        int M, int lda, int ldc, int relu){
  int tiles=((M+31)/32)*((L.N+31)/32);
  int blocks=(tiles+3)/4;                  // 4 waves (128 threads) per block
  gemm_wmma<<<blocks,128,0,st>>>(A,L.Wb,L.b,L.s,L.h,Cadd,C,M,L.N,L.Kp,lda,ldc,relu);
}

// relu(lbn2(relu(lbn1(X))) + lbn_dn(X)) -> out; t1 scratch [M, mid]
static void resblock_launch(hipStream_t st, const float* X, int M, int lda,
                            const RES& p, float* t1, float* out, int ldo){
  gemm_launch(st, X,  p.l1, nullptr, t1,  M, lda,      p.l1.N, 1);
  gemm_launch(st, X,  p.dn, nullptr, out, M, lda,      ldo,    0);
  gemm_launch(st, t1, p.l2, out,     out, M, p.l1.N,   ldo,    1);
}

extern "C" void kernel_launch(void* const* d_in, const int* in_sizes, int n_in,
                              void* d_out, int out_size, void* d_ws, size_t ws_size,
                              hipStream_t stream){
  (void)in_sizes; (void)n_in; (void)out_size; (void)ws_size;
  const float* atlas=(const float*)d_in[0];
  const float* feat =(const float*)d_in[1];
  const float* pos  =(const float*)d_in[2];
  // d_in[3] = batch (all zeros, unused)

  // --- params in setup_inputs() dict-insertion order ---
  int pi=4;
  auto nx=[&]()->const float*{ return (const float*)d_in[pi++]; };
  const float *tW1=nx(),*tb1=nx(),*ts1=nx(),*th1=nx();
  const float *tW2=nx(),*tb2=nx(),*ts2=nx(),*th2=nx();
  const float *tWl=nx(),*tbl=nx();
  auto nlbn=[&](int K,int N)->LBN{
    LBN l; l.W=nx(); l.b=nx(); l.s=nx(); l.h=nx();
    l.K=K; l.N=N; l.Kp=rup(K,32); l.Wb=nullptr; return l;
  };
  auto nres=[&](int c0,int c1,int c2)->RES{
    RES r; r.l1=nlbn(c0,c1); r.l2=nlbn(c1,c2); r.dn=nlbn(c0,c2); return r;
  };
  RES sa1a=nres(84,64,64),     sa1b=nres(84,64,64);
  RES sa2a=nres(131,256,256),  sa2b=nres(131,256,256);
  RES sa3 =nres(515,1024,1024);
  RES fp3 =nres(1536,512,512), fp2=nres(640,256,256), fp1=nres(337,128,128);
  LBN lin1, lin2, lin3;
  lin1.W=nx(); lin1.b=nx(); lin1.s=nullptr; lin1.h=nullptr; lin1.K=128; lin1.N=128; lin1.Kp=128;
  lin2.W=nx(); lin2.b=nx(); lin2.s=nullptr; lin2.h=nullptr; lin2.K=128; lin2.N=128; lin2.Kp=128;
  lin3.W=nx(); lin3.b=nx(); lin3.s=nullptr; lin3.h=nullptr; lin3.K=128; lin3.N=NCLS; lin3.Kp=128;

  // --- workspace bump allocation (256B-aligned) ---
  char* wsb=(char*)d_ws; size_t woff=0;
  auto alloc=[&](size_t bytes)->void*{
    woff=(woff+255)&~(size_t)255; void* p=wsb+woff; woff+=bytes; return p;
  };
  auto falloc=[&](size_t nf)->float*{ return (float*)alloc(nf*sizeof(float)); };

  // bf16 weight buffers + prep (Np padded to 32 for 32-wide N tiling)
  LBN* all_lbn[27]={
    &sa1a.l1,&sa1a.l2,&sa1a.dn,&sa1b.l1,&sa1b.l2,&sa1b.dn,
    &sa2a.l1,&sa2a.l2,&sa2a.dn,&sa2b.l1,&sa2b.l2,&sa2b.dn,
    &sa3.l1,&sa3.l2,&sa3.dn,
    &fp3.l1,&fp3.l2,&fp3.dn,
    &fp2.l1,&fp2.l2,&fp2.dn,
    &fp1.l1,&fp1.l2,&fp1.dn,
    &lin1,&lin2,&lin3 };
  for(int i=0;i<27;i++){
    LBN* L=all_lbn[i];
    int Np=rup(L->N,32);
    L->Wb=(__bf16*)alloc((size_t)Np*L->Kp*sizeof(__bf16));
    int tot=Np*L->Kp;
    prep_w<<<(tot+255)/256,256,0,stream>>>(L->W,L->K,L->N,L->Wb,L->Kp,Np);
  }

  // activations / state
  float* x0  =falloc((size_t)NPTS*81);
  float* pos1=falloc((size_t)NS1*3);
  float* pos2=falloc((size_t)NS2*3);
  float* x1  =falloc((size_t)NS1*128);
  float* x2  =falloc((size_t)NS2*512);
  float* x3  =falloc(1024);
  float* h2  =falloc((size_t)NS2*512);
  float* h1  =falloc((size_t)NS1*256);
  float* h0  =falloc((size_t)NPTS*128);
  float* fpsd=falloc(NPTS);
  int*   idx1=(int*)alloc(NS1*sizeof(int));
  int*   idx2=(int*)alloc(NS2*sizeof(int));
  int*   nbrF=(int*)alloc((size_t)NS1*192*sizeof(int));
  int*   nbrS=(int*)alloc((size_t)NS1*32*sizeof(int));
  int*   iidx=(int*)alloc((size_t)NPTS*3*sizeof(int));
  float* id2 =falloc((size_t)NPTS*3);
  // scratch arenas (widths padded to 32; row overhang stays in-arena)
  float* SA  =falloc((size_t)NS1*32*96);    // 7.55M floats: widest GEMM input
  float* SB  =falloc((size_t)NS1*32*64);    // t1 arena
  float* SC  =falloc((size_t)NS1*32*64);    // resblock output arena

  // ---- stage 0: texture CNN + feature concat -> x0 [N,81] ----
  tex_kernel<<<NPTS,128,0,stream>>>(atlas,feat,tW1,tb1,ts1,th1,tW2,tb2,ts2,th2,tWl,tbl,x0);

  const int Ks[2]={16,32}, dils[2]={3,6};

  // ---- SA1 -> x1 [NS1,128] ----
  fps_kernel<<<1,1024,0,stream>>>(pos,NPTS,NS1,idx1,fpsd);
  gather3<<<(NS1*3+255)/256,256,0,stream>>>(pos,idx1,NS1,pos1);
  const RES* sa1p[2]={&sa1a,&sa1b};
  for(int b=0;b<2;b++){
    int K=Ks[b], KD=K*dils[b];
    size_t smem=(size_t)NPTS*4+256*8;
    knn_block<<<NS1,256,smem,stream>>>(pos,NPTS,pos1,KD,nbrF);
    subsample_kernel<<<(NS1*K+255)/256,256,0,stream>>>(nbrF,NS1,K,KD,0x1000u+b,nbrS);
    build_msg<<<NS1*K,128,0,stream>>>(x0,81,81,pos,pos1,nbrS,K,SA,96);
    resblock_launch(stream,SA,NS1*K,96,*sa1p[b],SB,SC,64);
    max_aggr<<<NS1,64,0,stream>>>(SC,K,64,64,x1,128,b*64);
  }

  // ---- SA2 -> x2 [NS2,512] ----
  fps_kernel<<<1,1024,0,stream>>>(pos1,NS1,NS2,idx2,fpsd);
  gather3<<<(NS2*3+255)/256,256,0,stream>>>(pos1,idx2,NS2,pos2);
  const RES* sa2p[2]={&sa2a,&sa2b};
  for(int b=0;b<2;b++){
    int K=Ks[b], KD=K*dils[b];
    size_t smem=(size_t)NS1*4+256*8;
    knn_block<<<NS2,256,smem,stream>>>(pos1,NS1,pos2,KD,nbrF);
    subsample_kernel<<<(NS2*K+255)/256,256,0,stream>>>(nbrF,NS2,K,KD,0x2000u+b,nbrS);
    build_msg<<<NS2*K,128,0,stream>>>(x1,128,128,pos1,pos2,nbrS,K,SA,160);
    resblock_launch(stream,SA,NS2*K,160,*sa2p[b],SB,SC,256);
    max_aggr<<<NS2,256,0,stream>>>(SC,K,256,256,x2,512,b*256);
  }

  // ---- SA3: global resblock + max pool -> x3 [1,1024] ----
  concat2<<<NS2,256,0,stream>>>(x2,512,pos2,3,SA,544);
  resblock_launch(stream,SA,NS2,544,sa3,SB,SC,1024);
  colmax<<<(1024+255)/256,256,0,stream>>>(SC,NS2,1024,1024,x3);

  // ---- FP3: broadcast x3 + x2 -> resblock(1536->512) -> h2 ----
  interp_concat<<<NS2,256,0,stream>>>(x3,1024,1024,nullptr,nullptr,0,x2,512,512,SA,1536);
  resblock_launch(stream,SA,NS2,1536,fp3,SB,h2,512);

  // ---- FP2: knn3 interp h2 + x1 -> resblock(640->256) -> h1 ----
  knn3_kernel<<<(NS1+255)/256,256,0,stream>>>(pos2,NS2,pos1,NS1,iidx,id2);
  interp_concat<<<NS1,256,0,stream>>>(h2,512,512,iidx,id2,3,x1,128,128,SA,640);
  resblock_launch(stream,SA,NS1,640,fp2,SB,h1,256);

  // ---- FP1: knn3 interp h1 + x0 -> resblock(337->128) -> h0 ----
  knn3_kernel<<<(NPTS+255)/256,256,0,stream>>>(pos1,NS1,pos,NPTS,iidx,id2);
  interp_concat<<<NPTS,256,0,stream>>>(h1,256,256,iidx,id2,3,x0,81,81,SA,352);
  resblock_launch(stream,SA,NPTS,352,fp1,SB,h0,128);

  // ---- head: lin1(relu) -> lin2 -> lin3 -> log_softmax ----
  float* out=(float*)d_out;
  gemm_launch(stream,h0,lin1,nullptr,SB,NPTS,128,128,1);
  gemm_launch(stream,SB,lin2,nullptr,SC,NPTS,128,128,0);
  gemm_launch(stream,SC,lin3,nullptr,out,NPTS,128,NCLS,0);
  logsoftmax13<<<(NPTS+255)/256,256,0,stream>>>(out,NPTS);
}
